// BiLSTM_CRF_6597069766798
// MI455X (gfx1250) — matrix operand order repaired and checked by reading, not compiled, not run on hardware
//
#include <hip/hip_runtime.h>
#include <stdint.h>

#define T_DIM   4096
#define D_DIM   2048
#define HH_DIM  1024
#define G_DIM   4096    // 4*HH
#define NTAGS   32
#define START_TAG 30
#define STOP_TAG  31
#define NEG_VAL  (-10000.0f)

typedef __bf16 v16bf __attribute__((ext_vector_type(16)));
typedef __bf16 v8bf  __attribute__((ext_vector_type(8)));
typedef float  v8f   __attribute__((ext_vector_type(8)));
typedef unsigned int u32x4 __attribute__((ext_vector_type(4)));
typedef int          i32x8 __attribute__((ext_vector_type(8)));
typedef int          i32x4 __attribute__((ext_vector_type(4)));

// ---------------------------------------------------------------------------
// TDM helpers (CDNA5 ISA: D# group0/group1 bitfields)
// ---------------------------------------------------------------------------
__device__ __forceinline__ unsigned lds_off(const void* p) {
  // Generic LDS address: addr[31:0] is the LDS byte offset (ISA §10.2 aperture).
  return (unsigned)(uintptr_t)p;
}

// 2D tile load, data_size = 4 bytes; all dims/strides in dword units.
__device__ __forceinline__ void tdm_load_2d(unsigned ldsByteOff, const void* gptr,
                                            unsigned tensorW_dw, unsigned tensorH,
                                            unsigned tileW_dw,   unsigned tileH,
                                            unsigned rowStride_dw)
{
  unsigned long long ga = (unsigned long long)(uintptr_t)gptr;
  u32x4 g0;
  g0[0] = 1u;                                          // count=1 (valid), user mode
  g0[1] = ldsByteOff;                                  // lds_addr [63:32]
  g0[2] = (unsigned)(ga & 0xffffffffu);                // global_addr[31:0]
  g0[3] = (unsigned)((ga >> 32) & 0x01ffffffu)         // global_addr[56:32]
        | (2u << 30);                                  // type=2 ("image")
  i32x8 g1;
  g1[0] = (int)(2u << 16);                             // data_size=2 -> 4 bytes
  g1[1] = (int)((tensorW_dw & 0xffffu) << 16);         // tensor_dim0[15:0]  @ bits 63:48
  g1[2] = (int)(((tensorW_dw >> 16) & 0xffffu)         // tensor_dim0[31:16] @ bits 79:64
        | ((tensorH & 0xffffu) << 16));                // tensor_dim1[15:0]  @ bits 111:96? (95:80)
  g1[3] = (int)(((tensorH >> 16) & 0xffffu)            // tensor_dim1[31:16]
        | ((tileW_dw & 0xffffu) << 16));               // tile_dim0 @ bits 127:112
  g1[4] = (int)(tileH & 0xffffu);                      // tile_dim1 (tile_dim2=0)
  g1[5] = (int)rowStride_dw;                           // tensor_dim0_stride[31:0]
  g1[6] = 0;                                           // stride hi / dim1_stride lo
  g1[7] = 0;
  i32x4 z4; z4[0] = z4[1] = z4[2] = z4[3] = 0;
#if defined(__clang_major__) && (__clang_major__ >= 23)
  i32x8 z8; 
  #pragma unroll
  for (int i = 0; i < 8; ++i) z8[i] = 0;
  __builtin_amdgcn_tensor_load_to_lds(g0, g1, z4, z4, z8, 0);
#else
  __builtin_amdgcn_tensor_load_to_lds(g0, g1, z4, z4, 0);
#endif
}

// ---------------------------------------------------------------------------
// fp32 -> bf16 (RNE) conversion
// ---------------------------------------------------------------------------
__global__ void __launch_bounds__(256)
f32_to_bf16_k(const float* __restrict__ src, unsigned short* __restrict__ dst, size_t n)
{
  size_t i = (size_t)blockIdx.x * 256 + threadIdx.x;
  const size_t stride = (size_t)gridDim.x * 256;
  for (; i < n; i += stride) {
    unsigned u = __float_as_uint(src[i]);
    dst[i] = (unsigned short)((u + 0x7fffu + ((u >> 16) & 1u)) >> 16);
  }
}

__global__ void zero_sync_k(unsigned* s) {
  if (threadIdx.x < 16) s[threadIdx.x] = 0u;
}

// ---------------------------------------------------------------------------
// bf16 WMMA GEMM:  Y[t,g] = sum_k X[t,k]*W[g,k] + bias[g]
// Block tile 128x128, BK=32, 8 waves (4 M-groups x 2 N-groups),
// per wave: 2x4 = 8 accumulators of 16x16 -> 8 v_wmma per K-step.
// Tiles staged to LDS by the Tensor Data Mover, double buffered.
// ---------------------------------------------------------------------------
#define BM 128
#define BN 128
#define BK 32

__global__ void __launch_bounds__(256)
wmma_gemm_bf16(const unsigned short* __restrict__ X,   // [T_DIM x K] bf16 bits
               const unsigned short* __restrict__ W,   // [G_DIM x K] bf16 bits
               const float* __restrict__ bias,         // [G_DIM]
               float* __restrict__ Y,                  // [T_DIM x ldY]
               int K, int ldY)
{
  __shared__ __attribute__((aligned(64))) unsigned short As[2][BM * BK];
  __shared__ __attribute__((aligned(64))) unsigned short Bs[2][BN * BK];

  const int tid  = threadIdx.x;
  const int lane = tid & 31;
  const int wave = tid >> 5;       // 0..7
  const int wm   = wave >> 1;      // 0..3 : 32-row M group
  const int wn   = wave & 1;       // 0..1 : 64-col N group
  const int tileM = blockIdx.y * BM;
  const int tileN = blockIdx.x * BN;

  v8f acc[2][4];
  #pragma unroll
  for (int a = 0; a < 2; ++a)
    #pragma unroll
    for (int b = 0; b < 4; ++b)
      #pragma unroll
      for (int v = 0; v < 8; ++v) acc[a][b][v] = 0.0f;

  const unsigned rowDw = (unsigned)(K >> 1);    // bf16 row length in dwords

  // Prologue: stage k-block 0 into buffer 0 (wave 0 issues; TDM is wave-level)
  if (tid < 32) {
    tdm_load_2d(lds_off(&As[0][0]), X + (size_t)tileM * K, rowDw, (unsigned)T_DIM,
                BK / 2, BM, rowDw);
    tdm_load_2d(lds_off(&Bs[0][0]), W + (size_t)tileN * K, rowDw, (unsigned)G_DIM,
                BK / 2, BN, rowDw);
    __builtin_amdgcn_s_wait_tensorcnt(0);
  }
  __syncthreads();

  const int nk   = K / BK;
  const int half = lane >> 4;   // 0: lanes 0-15, 1: lanes 16-31
  const int l15  = lane & 15;

  for (int kk = 0; kk < nk; ++kk) {
    const int buf = kk & 1;
    if (kk + 1 < nk && tid < 32) {   // async prefetch of next K tile via TDM
      tdm_load_2d(lds_off(&As[buf ^ 1][0]), X + (size_t)tileM * K + (size_t)(kk + 1) * BK,
                  rowDw, (unsigned)T_DIM, BK / 2, BM, rowDw);
      tdm_load_2d(lds_off(&Bs[buf ^ 1][0]), W + (size_t)tileN * K + (size_t)(kk + 1) * BK,
                  rowDw, (unsigned)G_DIM, BK / 2, BN, rowDw);
    }

    // A fragments (16x32, contiguous K per lane half, same order as B):
    //   lanes 0-15 : row M=l15, K {0..15};  lanes 16-31: K {16..31}
    v16bf afrag[2], bfrag[4];
    #pragma unroll
    for (int fm = 0; fm < 2; ++fm) {
      const unsigned short* pa = &As[buf][(wm * 32 + fm * 16 + l15) * BK + half * 16];
      v8bf lo = *(const v8bf*)pa;
      v8bf hi = *(const v8bf*)(pa + 8);
      #pragma unroll
      for (int e = 0; e < 8; ++e) { afrag[fm][e] = lo[e]; afrag[fm][8 + e] = hi[e]; }
    }
    // B fragments (32x16): lane holds column n = l15, contiguous K half per lane-group
    #pragma unroll
    for (int fn = 0; fn < 4; ++fn) {
      const unsigned short* pb = &Bs[buf][(wn * 64 + fn * 16 + l15) * BK + half * 16];
      bfrag[fn] = *(const v16bf*)pb;
    }

    #pragma unroll
    for (int fm = 0; fm < 2; ++fm)
      #pragma unroll
      for (int fn = 0; fn < 4; ++fn)
        acc[fm][fn] = __builtin_amdgcn_wmma_f32_16x16x32_bf16(
            false, afrag[fm], false, bfrag[fn], (short)0, acc[fm][fn], false, false);

    if (kk + 1 < nk && tid < 32) __builtin_amdgcn_s_wait_tensorcnt(0);
    __syncthreads();
  }

  // C/D layout (ISA §7.12.2): VGPR v, lanes 0-15 -> M=v, lanes 16-31 -> M=v+8; N=l15
  #pragma unroll
  for (int fm = 0; fm < 2; ++fm) {
    #pragma unroll
    for (int fn = 0; fn < 4; ++fn) {
      const int col = tileN + wn * 64 + fn * 16 + l15;
      const float bv = bias[col];
      #pragma unroll
      for (int v = 0; v < 8; ++v) {
        const int row = tileM + wm * 32 + fm * 16 + half * 8 + v;
        Y[(size_t)row * ldY + col] = acc[fm][fn][v] + bv;
      }
    }
  }
}

// ---------------------------------------------------------------------------
// Persistent sequential LSTM: 2 directions x 256 blocks; each block pins its
// 16-row W_hh slice (bf16, 32 KB) in LDS via TDM, then runs the 4096-step
// scan with grid-wide split barriers (atomic release/acquire + s_sleep).
// ---------------------------------------------------------------------------
#define RB_ROWS 16
#define NBLK    256

__device__ __forceinline__ float bf2f(unsigned short h) {
  return __uint_as_float(((unsigned)h) << 16);
}
__device__ __forceinline__ float sigmoidf_(float x) { return 1.0f / (1.0f + __expf(-x)); }

__device__ void grid_barrier(volatile unsigned* cnt, volatile unsigned* gen, unsigned n) {
  __syncthreads();
  if (threadIdx.x == 0) {
    __threadfence();
    unsigned g = *gen;
    if (atomicAdd((unsigned*)cnt, 1u) == n - 1u) {
      *cnt = 0u;
      __threadfence();
      atomicAdd((unsigned*)gen, 1u);
    } else {
      while (*gen == g) { __builtin_amdgcn_s_sleep(1); }
    }
    __threadfence();
  }
  __syncthreads();
}

__global__ void __launch_bounds__(256)
lstm_seq(const unsigned short* __restrict__ WhhF, const unsigned short* __restrict__ WhhB,
         const float* __restrict__ xwF, const float* __restrict__ xwB,
         const float* __restrict__ bhhF, const float* __restrict__ bhhB,
         const float* __restrict__ h0, const float* __restrict__ c0,
         float* __restrict__ houtF, float* __restrict__ houtB,
         float* __restrict__ ggates, float* __restrict__ ghcur,
         float* __restrict__ gcell, unsigned* __restrict__ sync)
{
  const int dir = blockIdx.x / NBLK;   // 0 = forward, 1 = backward
  const int blk = blockIdx.x % NBLK;
  const unsigned short* Whh = dir ? WhhB : WhhF;
  const float* xw  = dir ? xwB  : xwF;
  const float* bhh = dir ? bhhB : bhhF;
  float* hout = dir ? houtB : houtF;
  float* gg = ggates + dir * G_DIM;
  float* hc = ghcur  + dir * HH_DIM;
  float* cc = gcell  + dir * HH_DIM;
  volatile unsigned* cnt = sync + dir * 2;
  volatile unsigned* gen = sync + dir * 2 + 1;

  __shared__ __attribute__((aligned(64))) unsigned short wl[RB_ROWS * HH_DIM]; // 32 KB
  __shared__ float hl[HH_DIM];
  __shared__ float part[RB_ROWS][17];

  const int tid = threadIdx.x;
  const int grow0 = blk * RB_ROWS;

  // Pin this block's W_hh slice in LDS (one TDM transfer, stays resident 4096 steps)
  if (tid < 32) {
    tdm_load_2d(lds_off(&wl[0]), Whh + (size_t)grow0 * HH_DIM,
                HH_DIM / 2, (unsigned)G_DIM, HH_DIM / 2, RB_ROWS, HH_DIM / 2);
    __builtin_amdgcn_s_wait_tensorcnt(0);
  }
  // Init recurrent state (each block owns 4 hidden units)
  if (tid < 4) {
    const int j = blk * 4 + tid;
    hc[j] = h0[dir * HH_DIM + j];
    cc[j] = c0[dir * HH_DIM + j];
  }
  grid_barrier(cnt, gen, NBLK);

  const int r = tid >> 4;       // gate row within slice (0..15)
  const int sub = tid & 15;     // 16-way K split per row

  for (int t = 0; t < T_DIM; ++t) {
    const int idx = (dir == 0) ? t : (T_DIM - 1 - t);

    // broadcast h into LDS
    const volatile float* hcv = hc;
    for (int k = tid; k < HH_DIM; k += 256) hl[k] = hcv[k];
    __syncthreads();

    // phase 1: raw gate pre-activations for this block's 16 gate rows
    float s = 0.0f;
    const unsigned short* wr = &wl[r * HH_DIM + sub * 64];
    const float* hp = &hl[sub * 64];
    #pragma unroll 8
    for (int k = 0; k < 64; ++k) s += bf2f(wr[k]) * hp[k];
    part[r][sub] = s;
    __syncthreads();
    if (sub == 0) {
      float a = 0.0f;
      #pragma unroll
      for (int i = 0; i < 16; ++i) a += part[r][i];
      const int gi = grow0 + r;
      gg[gi] = a + xw[(size_t)idx * G_DIM + gi] + bhh[gi];
    }
    grid_barrier(cnt, gen, NBLK);

    // phase 2: elementwise cell/hidden update for this block's 4 units
    if (tid < 4) {
      const int j = blk * 4 + tid;
      volatile float* ggv = gg;
      const float iv = sigmoidf_(ggv[j]);
      const float fv = sigmoidf_(ggv[HH_DIM + j]);
      const float gv = tanhf(ggv[2 * HH_DIM + j]);
      const float ov = sigmoidf_(ggv[3 * HH_DIM + j]);
      const float c  = fv * cc[j] + iv * gv;
      cc[j] = c;
      const float h = ov * tanhf(c);
      hc[j] = h;
      hout[(size_t)idx * HH_DIM + j] = h;
    }
    grid_barrier(cnt, gen, NBLK);
  }
}

// ---------------------------------------------------------------------------
// feats[t, n] = concat(h_f[t], h_b[t]) . W_out[n] + b_out[n]
// ---------------------------------------------------------------------------
__global__ void __launch_bounds__(256)
feats_k(const float* __restrict__ hf, const float* __restrict__ hb,
        const float* __restrict__ Wout, const float* __restrict__ bout,
        float* __restrict__ feats)
{
  const int t = blockIdx.x;
  const int tid = threadIdx.x;
  const int tag = tid >> 3;      // 32 tags
  const int sub = tid & 7;       // 8-way K split
  __shared__ float p[NTAGS][9];
  const float* wrow = Wout + (size_t)tag * (2 * HH_DIM);
  float s = 0.0f;
  const int k0 = sub * 128;
  for (int k = k0; k < k0 + 128; ++k) s += hf[(size_t)t * HH_DIM + k] * wrow[k];
  for (int k = k0; k < k0 + 128; ++k) s += hb[(size_t)t * HH_DIM + k] * wrow[HH_DIM + k];
  p[tag][sub] = s;
  __syncthreads();
  if (sub == 0) {
    float a = 0.0f;
    #pragma unroll
    for (int i = 0; i < 8; ++i) a += p[tag][i];
    feats[(size_t)t * NTAGS + tag] = a + bout[tag];
  }
}

// ---------------------------------------------------------------------------
// Viterbi forward scan: one wave32, lane = next-tag
// ---------------------------------------------------------------------------
__global__ void viterbi_fwd_k(const float* __restrict__ feats,
                              const float* __restrict__ trans,
                              float* __restrict__ fv_out, int* __restrict__ bptr)
{
  const int lane = threadIdx.x;    // 0..31
  float tr[NTAGS];
  #pragma unroll
  for (int p = 0; p < NTAGS; ++p) tr[p] = trans[lane * NTAGS + p];
  __shared__ float fv[NTAGS];
  fv[lane] = (lane == START_TAG) ? 0.0f : NEG_VAL;
  __syncthreads();
  for (int t = 0; t < T_DIM; ++t) {
    float best = -3.0e38f; int arg = 0;
    #pragma unroll
    for (int p = 0; p < NTAGS; ++p) {
      const float sc = fv[p] + tr[p];
      if (sc > best) { best = sc; arg = p; }
    }
    const float nf = best + feats[(size_t)t * NTAGS + lane];
    bptr[(size_t)t * NTAGS + lane] = arg;
    __syncthreads();
    fv[lane] = nf;
    __syncthreads();
  }
  fv_out[lane] = fv[lane];
}

__global__ void backtrace_k(const float* __restrict__ fv, const float* __restrict__ trans,
                            const int* __restrict__ bptr,
                            float* __restrict__ out_score, int* __restrict__ out_path)
{
  if (threadIdx.x == 0 && blockIdx.x == 0) {
    float best = -3.0e38f; int bt = 0;
    for (int j = 0; j < NTAGS; ++j) {
      const float s = fv[j] + trans[STOP_TAG * NTAGS + j];
      if (s > best) { best = s; bt = j; }
    }
    out_score[0] = best;
    int tag = bt;
    for (int t = T_DIM - 1; t >= 0; --t) {
      out_path[t] = tag;
      tag = bptr[(size_t)t * NTAGS + tag];
    }
  }
}

// ---------------------------------------------------------------------------
// Orchestration
// ---------------------------------------------------------------------------
extern "C" void kernel_launch(void* const* d_in, const int* in_sizes, int n_in,
                              void* d_out, int out_size, void* d_ws, size_t ws_size,
                              hipStream_t stream)
{
  (void)in_sizes; (void)n_in; (void)out_size; (void)ws_size;
  const float* embeds = (const float*)d_in[0];
  const float* W_ih_f = (const float*)d_in[1];
  const float* W_hh_f = (const float*)d_in[2];
  const float* b_ih_f = (const float*)d_in[3];
  const float* b_hh_f = (const float*)d_in[4];
  const float* W_ih_b = (const float*)d_in[5];
  const float* W_hh_b = (const float*)d_in[6];
  const float* b_ih_b = (const float*)d_in[7];
  const float* b_hh_b = (const float*)d_in[8];
  const float* W_out  = (const float*)d_in[9];
  const float* b_out  = (const float*)d_in[10];
  const float* trans  = (const float*)d_in[11];
  const float* h0     = (const float*)d_in[12];
  const float* c0     = (const float*)d_in[13];

  char* ws = (char*)d_ws;
  size_t off = 0;
  auto alloc = [&](size_t bytes) -> char* {
    char* p = ws + off;
    off = (off + bytes + 255) & ~(size_t)255;
    return p;
  };
  unsigned short* embX = (unsigned short*)alloc((size_t)T_DIM * D_DIM * 2);
  unsigned short* wihf = (unsigned short*)alloc((size_t)G_DIM * D_DIM * 2);
  unsigned short* wihb = (unsigned short*)alloc((size_t)G_DIM * D_DIM * 2);
  unsigned short* whhf = (unsigned short*)alloc((size_t)G_DIM * HH_DIM * 2);
  unsigned short* whhb = (unsigned short*)alloc((size_t)G_DIM * HH_DIM * 2);
  float* xwf   = (float*)alloc((size_t)T_DIM * G_DIM * 4);
  float* xwb   = (float*)alloc((size_t)T_DIM * G_DIM * 4);
  float* hf    = (float*)alloc((size_t)T_DIM * HH_DIM * 4);
  float* hb    = (float*)alloc((size_t)T_DIM * HH_DIM * 4);
  float* gg    = (float*)alloc(2 * G_DIM * 4);
  float* hc    = (float*)alloc(2 * HH_DIM * 4);
  float* cc    = (float*)alloc(2 * HH_DIM * 4);
  float* feats = (float*)alloc((size_t)T_DIM * NTAGS * 4);
  int*   bptr  = (int*)alloc((size_t)T_DIM * NTAGS * 4);
  float* fvf   = (float*)alloc(NTAGS * 4);
  unsigned* syncws = (unsigned*)alloc(64);

  zero_sync_k<<<1, 32, 0, stream>>>(syncws);

  f32_to_bf16_k<<<2048, 256, 0, stream>>>(embeds, embX, (size_t)T_DIM * D_DIM);
  f32_to_bf16_k<<<2048, 256, 0, stream>>>(W_ih_f, wihf, (size_t)G_DIM * D_DIM);
  f32_to_bf16_k<<<2048, 256, 0, stream>>>(W_ih_b, wihb, (size_t)G_DIM * D_DIM);
  f32_to_bf16_k<<<1024, 256, 0, stream>>>(W_hh_f, whhf, (size_t)G_DIM * HH_DIM);
  f32_to_bf16_k<<<1024, 256, 0, stream>>>(W_hh_b, whhb, (size_t)G_DIM * HH_DIM);

  dim3 ggrid(G_DIM / BN, T_DIM / BM);   // 32 x 32 blocks
  wmma_gemm_bf16<<<ggrid, 256, 0, stream>>>(embX, wihf, b_ih_f, xwf, D_DIM, G_DIM);
  wmma_gemm_bf16<<<ggrid, 256, 0, stream>>>(embX, wihb, b_ih_b, xwb, D_DIM, G_DIM);

  lstm_seq<<<2 * NBLK, 256, 0, stream>>>(whhf, whhb, xwf, xwb, b_hh_f, b_hh_b,
                                         h0, c0, hf, hb, gg, hc, cc, syncws);

  feats_k<<<T_DIM, 256, 0, stream>>>(hf, hb, W_out, b_out, feats);
  viterbi_fwd_k<<<1, 32, 0, stream>>>(feats, trans, fvf, bptr);
  backtrace_k<<<1, 32, 0, stream>>>(fvf, trans, bptr, (float*)d_out, (int*)d_out + 1);
}
